// MyRNN_76416058131306
// MI455X (gfx1250) — matrix operand order repaired and checked by reading, not compiled, not run on hardware
//
#include <hip/hip_runtime.h>
#include <hip/hip_bf16.h>

// ---------------------------------------------------------------------------
// RNN forward for MI455X (gfx1250, wave32, WMMA + TDM + async-LDS).
//   Phase 1: xw = x @ W_ih + b_ih + b_hh  -> d_out
//            W_ih slab staged into LDS by the Tensor Data Mover (one D#),
//            bf16 WMMA with fp32 accumulate.
//   Phase 2: persistent 16-block kernel, 1024 sequential steps:
//            h staged into LDS with global_load_async_to_lds_b128 each step,
//            W_hh held in VGPRs (bf16) per wave, grid barrier per step.
// ---------------------------------------------------------------------------

typedef __attribute__((ext_vector_type(16))) __bf16 v16bf;
typedef __attribute__((ext_vector_type(8)))  __bf16 v8bf;
typedef __attribute__((ext_vector_type(8)))  float  v8f;

#define BATCH 64
#define SEQ   1024
#define ISZ   512
#define HSZ   512
#define RNN_BLOCKS 16
#define M_PER_BLOCK 256
#define HBUF_BYTES (BATCH * HSZ * 2)           // bf16 hidden state
#define HN_OFF     ((size_t)BATCH * SEQ * HSZ) // hn location in d_out

__device__ __forceinline__ float dev_tanh(float x) {
#if __has_builtin(__builtin_amdgcn_tanhf)
  return __builtin_amdgcn_tanhf(x);   // V_TANH_F32 (gfx1250 trans op)
#else
  return tanhf(x);
#endif
}

// ---------------------------------------------------------------------------
// Tensor Data Mover: 2D tile (fp32) global -> LDS.  D# per ISA 08 §8.
// tile_d0 = contiguous elements per row, tile_d1 = rows, stride0 = row stride.
// ---------------------------------------------------------------------------
typedef __attribute__((ext_vector_type(4))) unsigned tdm_v4u;
typedef __attribute__((ext_vector_type(8))) int      tdm_v8i;
typedef __attribute__((ext_vector_type(4))) int      tdm_v4i;

__device__ __forceinline__ void tdm_load_2d_f32(
    unsigned lds_addr, const void* gaddr, int tensor_d0, int tensor_d1,
    int tile_d0, int tile_d1, int stride0) {
  unsigned long long ga = (unsigned long long)gaddr;
  tdm_v4u g0 = {
      1u,                                               // count=1 (valid user D#)
      lds_addr,                                         // LDS dest (bytes)
      (unsigned)ga,                                     // global_addr[31:0]
      (unsigned)((ga >> 32) & 0x01FFFFFFu) | (2u << 30) // addr[56:32] | type=2
  };
  tdm_v8i g1 = {
      (int)(2u << 16),                       // data_size=2 (4 bytes)
      (int)((unsigned)tensor_d0 << 16),      // tensor_dim0[15:0] (dims < 64K)
      (int)((unsigned)tensor_d1 << 16),      // tensor_dim0[31:16]=0 | dim1[15:0]
      (int)((unsigned)tile_d0 << 16),        // tensor_dim1[31:16]=0 | tile_dim0
      tile_d1,                               // tile_dim1 | tile_dim2=0
      stride0,                               // tensor_dim0_stride[31:0]
      0, 0                                   // stride hi / dim1_stride
  };
  tdm_v4i gz = {0, 0, 0, 0};
#if defined(__clang_major__) && __clang_major__ >= 23
  tdm_v8i z8 = {0, 0, 0, 0, 0, 0, 0, 0};
  __builtin_amdgcn_tensor_load_to_lds(g0, g1, gz, gz, z8, 0);
#else
  __builtin_amdgcn_tensor_load_to_lds(g0, g1, gz, gz, 0);
#endif
}

__device__ __forceinline__ unsigned lds_off(const void* p) {
  return (unsigned)(unsigned long long)p;  // AS3 generic addr low 32b = LDS offset
}

// ---------------------------------------------------------------------------
// Phase 1: xw[m, n] = sum_k x[m,k] * W_ih[k,n] + b_ih[n] + b_hh[n]
// Grid: (M/256, H/32), 256 threads (8 waves). Wave -> 4 M-tiles x 1 N-tile.
// ---------------------------------------------------------------------------
__global__ __launch_bounds__(256) void xw_kernel(
    const float* __restrict__ x, const float* __restrict__ Wih,
    const float* __restrict__ bih, const float* __restrict__ bhh,
    float* __restrict__ out)
{
  __shared__ __align__(16) float sW[ISZ * 32];   // 64KB: W_ih[0:512, c0:c0+32]

  const int lane = threadIdx.x & 31;
  const int wave = threadIdx.x >> 5;
  const int half = lane >> 4;
  const int ln   = lane & 15;
  const int c0blk = blockIdx.y * 32;

  // --- TDM: DMA the whole fp32 weight slab into LDS with one descriptor ---
  if (wave == 0) {
    tdm_load_2d_f32(lds_off(&sW[0]), Wih + c0blk, /*tensor_d0=*/HSZ,
                    /*tensor_d1=*/ISZ, /*tile_d0=*/32, /*tile_d1=*/ISZ,
                    /*stride0=*/HSZ);
    __builtin_amdgcn_s_wait_tensorcnt(0);
  }
  __syncthreads();

  // B fragments (bf16) from LDS. Element e of chunk c: K = 32c + 16*half + e.
  const int cl = (wave >> 2) * 16 + ln;          // block-local column
  v16bf bfrag[16];
#pragma unroll
  for (int c = 0; c < 16; ++c) {
#pragma unroll
    for (int e = 0; e < 16; ++e)
      bfrag[c][e] = (__bf16)sW[(c * 32 + half * 16 + e) * 32 + cl];
  }

  const int col = c0blk + cl;
  const float bias = bih[col] + bhh[col];

#pragma unroll 1
  for (int mt = 0; mt < 4; ++mt) {
    const int m0 = blockIdx.x * M_PER_BLOCK + (wave & 3) * 16 + mt * 64;
    const float* xrow = x + (size_t)(m0 + ln) * ISZ;
    v8f acc = {};
#pragma unroll
    for (int c = 0; c < 16; ++c) {
      const float* p0 = xrow + c * 32 + half * 16; // K = 16*half + 0..7
      const float* p1 = p0 + 8;                    // K = 16*half + 8..15
      v16bf a;
#pragma unroll
      for (int e = 0; e < 8; ++e) {
        a[e]     = (__bf16)p0[e];
        a[e + 8] = (__bf16)p1[e];
      }
      acc = __builtin_amdgcn_wmma_f32_16x16x32_bf16(
          false, a, false, bfrag[c], (short)0, acc, false, false);
    }
#pragma unroll
    for (int r = 0; r < 8; ++r) {                  // C/D: M = m0+8*half+r, N = col
      const int m = m0 + half * 8 + r;
      out[(size_t)m * HSZ + col] = acc[r] + bias;
    }
  }
}

// ---------------------------------------------------------------------------
// Phase 2: persistent recurrence. 16 blocks x 256 threads.
// Each wave keeps its 512x16 bf16 W_hh slice in VGPRs for all 1024 steps.
// h (64KB bf16) is staged global->LDS each step via async-to-LDS loads.
// ---------------------------------------------------------------------------
__global__ __launch_bounds__(256) void rnn_kernel(
    const float* __restrict__ Whh, float* __restrict__ out,
    __bf16* __restrict__ hbuf, unsigned* __restrict__ bar)
{
  __shared__ __align__(16) __bf16 sh[BATCH * HSZ];   // 64KB h mirror

  const int lane = threadIdx.x & 31;
  const int wave = threadIdx.x >> 5;
  const int half = lane >> 4;
  const int ln   = lane & 15;
  const int m0 = (wave & 3) * 16;                    // batch-row tile base
  const int c0 = blockIdx.x * 32 + (wave >> 2) * 16; // hidden-col tile base
  const int col = c0 + ln;

  // Persistent B = W_hh[:, c0:c0+16] as bf16 fragments (128 VGPRs).
  v16bf bfrag[16];
#pragma unroll
  for (int c = 0; c < 16; ++c) {
    const float* wp = Whh + (size_t)(c * 32 + half * 16) * HSZ + col;
#pragma unroll
    for (int e = 0; e < 16; ++e) bfrag[c][e] = (__bf16)wp[(size_t)e * HSZ];
  }

  const unsigned lbase = lds_off(&sh[0]);
  float lastv[8];

  for (int t = 0; t < SEQ; ++t) {
    // ---- stage h into LDS: 256 thr x 16 x b128 async copies (ASYNCcnt) ----
    {
      const char* gsrc = (const char*)hbuf;
#pragma unroll
      for (int i = 0; i < 16; ++i) {
        const unsigned off = ((unsigned)threadIdx.x + i * 256u) * 16u;
        const unsigned ldst = lbase + off;
        const unsigned long long g = (unsigned long long)(gsrc + off);
        asm volatile("global_load_async_to_lds_b128 %0, %1, off"
                     :: "v"(ldst), "v"(g) : "memory");
      }
      asm volatile("s_wait_asynccnt 0x0" ::: "memory");
    }
    __syncthreads();

    const __bf16* hrow = sh + (size_t)(m0 + ln) * HSZ;
    v8f acc = {};
#pragma unroll
    for (int c = 0; c < 16; ++c) {
      v8bf alo = *(const v8bf*)(hrow + c * 32 + half * 16);      // ds_load_b128
      v8bf ahi = *(const v8bf*)(hrow + c * 32 + half * 16 + 8);  // ds_load_b128
      v16bf a = __builtin_shufflevector(alo, ahi, 0, 1, 2, 3, 4, 5, 6, 7,
                                        8, 9, 10, 11, 12, 13, 14, 15);
      acc = __builtin_amdgcn_wmma_f32_16x16x32_bf16(
          false, a, false, bfrag[c], (short)0, acc, false, false);
    }

    // h = tanh(xw_t + acc); xw_t lives in (and is overwritten in) d_out.
#pragma unroll
    for (int r = 0; r < 8; ++r) {
      const int b = m0 + half * 8 + r;
      const size_t idx = ((size_t)b * SEQ + t) * HSZ + col;
      float v = dev_tanh(acc[r] + out[idx]);
      out[idx] = v;                               // output[b, t, :]
      hbuf[(size_t)b * HSZ + col] = (__bf16)v;    // next step's A operand
      lastv[r] = v;
    }

    // ---- device-scope grid barrier (sequential dependence across blocks) ----
    __builtin_amdgcn_fence(__ATOMIC_RELEASE, "agent");  // publish h stores
    __syncthreads();
    if (threadIdx.x == 0) {
      atomicAdd(bar, 1u);
      const unsigned target = (unsigned)gridDim.x * (unsigned)(t + 1);
      while (__hip_atomic_load(bar, __ATOMIC_RELAXED,
                               __HIP_MEMORY_SCOPE_AGENT) < target)
        __builtin_amdgcn_s_sleep(1);
    }
    __syncthreads();
    __builtin_amdgcn_fence(__ATOMIC_ACQUIRE, "agent");  // see others' h stores
  }

  // hn = h_{S-1} (fp32, exact copy of last output row)
#pragma unroll
  for (int r = 0; r < 8; ++r) {
    const int b = m0 + half * 8 + r;
    out[HN_OFF + (size_t)b * HSZ + col] = lastv[r];
  }
}

// ---------------------------------------------------------------------------
extern "C" void kernel_launch(void* const* d_in, const int* in_sizes, int n_in,
                              void* d_out, int out_size, void* d_ws,
                              size_t ws_size, hipStream_t stream) {
  const float* x   = (const float*)d_in[0];
  const float* Wih = (const float*)d_in[1];
  const float* bih = (const float*)d_in[2];
  const float* Whh = (const float*)d_in[3];
  const float* bhh = (const float*)d_in[4];
  float* out = (float*)d_out;

  __bf16*   hbuf = (__bf16*)d_ws;
  unsigned* bar  = (unsigned*)((char*)d_ws + HBUF_BYTES);

  // zero hidden state (h0 = 0) and barrier counter each launch (capture-safe)
  hipMemsetAsync(d_ws, 0, HBUF_BYTES + 64, stream);

  dim3 g1(BATCH * SEQ / M_PER_BLOCK, HSZ / 32);
  xw_kernel<<<g1, 256, 0, stream>>>(x, Wih, bih, bhh, out);
  rnn_kernel<<<RNN_BLOCKS, 256, 0, stream>>>(Whh, out, hbuf, bar);
}
